// MultiLayerLstm_7756710936904
// MI455X (gfx1250) — matrix operand order repaired and checked by reading, not compiled, not run on hardware
//
#include <hip/hip_runtime.h>

// ---------------------------------------------------------------------------
// CDNA5 (gfx1250) multi-layer LSTM forward. wave32, WMMA bf16 16x16x32.
// ---------------------------------------------------------------------------

typedef __attribute__((ext_vector_type(16))) __bf16 v16bf;
typedef __attribute__((ext_vector_type(8)))  __bf16 v8bf;
typedef __attribute__((ext_vector_type(8)))  float  v8f;

static __device__ __forceinline__ __bf16 f2bf(float f) {
    union { float f; unsigned u; } v; v.f = f;
    unsigned r = v.u + 0x7FFFu + ((v.u >> 16) & 1u);   // round-to-nearest-even
    unsigned short h = (unsigned short)(r >> 16);
    return __builtin_bit_cast(__bf16, h);
}

// A-fragment (16x32 bf16, row-major source, one row per lane-half):
// half==0: slots0-15 = K kk+0..15
// half==1: slots0-15 = K kk+16..31
static __device__ __forceinline__ v16bf load_a_frag(const __bf16* __restrict__ row,
                                                    int kk, int hv) {
    const v8bf lo = *(const v8bf*)(row + kk + 16 * hv);
    const v8bf hi = *(const v8bf*)(row + kk + 16 * hv + 8);
    v16bf a;
#pragma unroll
    for (int i = 0; i < 8; ++i) { a[i] = lo[i]; a[i + 8] = hi[i]; }
    return a;
}

// B-fragment from W^T storage: lane owns column n = n0 + (lane&15),
// K = kk + 16*half + (0..15)  -> contiguous 32B load from the W^T row.
static __device__ __forceinline__ v16bf load_b_frag(const __bf16* __restrict__ rowT,
                                                    int kk, int hv) {
    return *(const v16bf*)(rowT + kk + 16 * hv);
}

static __device__ __forceinline__ v8f wmma_bf16(v16bf a, v16bf b, v8f c) {
    return __builtin_amdgcn_wmma_f32_16x16x32_bf16(false, a, false, b,
                                                   (short)0, c, false, false);
}

// ---------------------------------------------------------------------------
// Generic GEMM: C[M][N] = A[M][K] (bf16, row-major) * Bt[N][K]^T (bf16) + bias
// Workgroup = 256 threads (8 waves) covering a 128x64 output tile.
// Wave w computes a 32x32 sub-tile: m at (w>>1)*32, n at (w&1)*32.
// 2 A-frags x 2 B-frags -> 4 WMMAs per 8 x b128 loads (2 loads/WMMA).
// ---------------------------------------------------------------------------
template <bool RELU, bool BF16OUT>
__global__ __launch_bounds__(256) void gemm_bf16_k(
    const __bf16* __restrict__ A, const __bf16* __restrict__ Bt,
    const float* __restrict__ bias, void* __restrict__ Cout,
    int M, int N, int K) {
    const int lane = threadIdx.x & 31;
    const int wave = threadIdx.x >> 5;
    const int hv   = lane >> 4;
    const int r    = lane & 15;

    const int m0 = blockIdx.y * 128 + (wave >> 1) * 32;
    const int n0 = blockIdx.x * 64 + (wave & 1) * 32;
    if (n0 >= N) return;                 // wave-uniform guard (N=32 case)
    const bool doN1 = (n0 + 16) < N;

    const __bf16* arow0 = A  + (size_t)(m0 + r) * K;
    const __bf16* arow1 = A  + (size_t)(m0 + 16 + r) * K;
    const __bf16* brow0 = Bt + (size_t)(n0 + r) * K;
    const __bf16* brow1 = Bt + (size_t)(n0 + 16 + r) * K;

    v8f acc00 = {}; v8f acc01 = {}; v8f acc10 = {}; v8f acc11 = {};
    for (int kk = 0; kk < K; kk += 32) {
        const v16bf a0 = load_a_frag(arow0, kk, hv);
        const v16bf a1 = load_a_frag(arow1, kk, hv);
        const v16bf b0 = load_b_frag(brow0, kk, hv);
        acc00 = wmma_bf16(a0, b0, acc00);
        acc10 = wmma_bf16(a1, b0, acc10);
        if (doN1) {
            const v16bf b1 = load_b_frag(brow1, kk, hv);
            acc01 = wmma_bf16(a0, b1, acc01);
            acc11 = wmma_bf16(a1, b1, acc11);
        }
    }

    const float bia0 = bias[n0 + r];
    const float bia1 = doN1 ? bias[n0 + 16 + r] : 0.0f;
#pragma unroll
    for (int ti = 0; ti < 2; ++ti) {
#pragma unroll
        for (int jj = 0; jj < 8; ++jj) {
            const int row = m0 + ti * 16 + jj + 8 * hv;
            float v0 = (ti ? acc10[jj] : acc00[jj]) + bia0;
            if (RELU) v0 = fmaxf(v0, 0.0f);
            const size_t i0 = (size_t)row * N + n0 + r;
            if (BF16OUT) ((__bf16*)Cout)[i0] = f2bf(v0);
            else         ((float*)Cout)[i0]  = v0;
            if (doN1) {
                float v1 = (ti ? acc11[jj] : acc01[jj]) + bia1;
                if (RELU) v1 = fmaxf(v1, 0.0f);
                const size_t i1 = (size_t)row * N + n0 + 16 + r;
                if (BF16OUT) ((__bf16*)Cout)[i1] = f2bf(v1);
                else         ((float*)Cout)[i1]  = v1;
            }
        }
    }
}

// ---------------------------------------------------------------------------
// LayerNorm(+ReLU) over rows of P=1024, one block (256 threads) per row.
// ---------------------------------------------------------------------------
__global__ __launch_bounds__(256) void ln_relu_k(
    const float* __restrict__ Z, const float* __restrict__ gamma,
    const float* __restrict__ beta, __bf16* __restrict__ out) {
    const int P = 1024;
    const int row = blockIdx.x;
    const float* z = Z + (size_t)row * P;
    float x[4], s = 0.f, q = 0.f;
#pragma unroll
    for (int i = 0; i < 4; ++i) {
        x[i] = z[threadIdx.x + 256 * i];
        s += x[i]; q += x[i] * x[i];
    }
    __shared__ float rs[256], rq[256];
    rs[threadIdx.x] = s; rq[threadIdx.x] = q;
    __syncthreads();
    for (int off = 128; off > 0; off >>= 1) {
        if (threadIdx.x < off) {
            rs[threadIdx.x] += rs[threadIdx.x + off];
            rq[threadIdx.x] += rq[threadIdx.x + off];
        }
        __syncthreads();
    }
    const float mu   = rs[0] * (1.0f / P);
    const float var  = rq[0] * (1.0f / P) - mu * mu;
    const float rstd = rsqrtf(var + 1e-6f);
#pragma unroll
    for (int i = 0; i < 4; ++i) {
        const int col = threadIdx.x + 256 * i;
        float v = (x[i] - mu) * rstd * gamma[col] + beta[col];
        out[(size_t)row * P + col] = f2bf(fmaxf(v, 0.0f));
    }
}

// ---------------------------------------------------------------------------
// One LSTM timestep. Grid = H/16 workgroups; each owns 16 H-columns.
// z[:, {i,f,g,o} cols] = hprev(64x1024) @ WhT + G[b*T+t]; then cell update.
// 8 waves: wave w -> gate = w>>1, two M-tiles mt = (w&1)*2, (w&1)*2+1,
// sharing the B fragment across both M-tiles.
// ---------------------------------------------------------------------------
__global__ __launch_bounds__(256) void lstm_step_k(
    const float*  __restrict__ G,       // [B*T][4H], bias already folded in
    const __bf16* __restrict__ WhT,     // [4H][H]
    float*        __restrict__ cstate,  // [B][H]
    const __bf16* __restrict__ hprev,   // [B][H]
    __bf16*       __restrict__ hnext,   // [B][H]
    __bf16*       __restrict__ hseq,    // [B*T][H]
    float* __restrict__ c_out, float* __restrict__ h_out,
    int t, int B, int T, int H) {
    const int lane = threadIdx.x & 31;
    const int wave = threadIdx.x >> 5;
    const int hv   = lane >> 4;
    const int r    = lane & 15;
    const int n0   = blockIdx.x * 16;
    const int gate = wave >> 1;
    const int mt0  = (wave & 1) * 2;

    const __bf16* brow = WhT + (size_t)(gate * H + n0 + r) * H;
    const __bf16* a0   = hprev + (size_t)(mt0 * 16 + r) * H;
    const __bf16* a1   = hprev + (size_t)(mt0 * 16 + 16 + r) * H;

    v8f acc0 = {}; v8f acc1 = {};
    for (int kk = 0; kk < H; kk += 32) {
        const v16bf bf = load_b_frag(brow, kk, hv);
        const v16bf f0 = load_a_frag(a0, kk, hv);
        const v16bf f1 = load_a_frag(a1, kk, hv);
        acc0 = wmma_bf16(f0, bf, acc0);
        acc1 = wmma_bf16(f1, bf, acc1);
    }

    __shared__ float zb[4][64][16];
#pragma unroll
    for (int pp = 0; pp < 2; ++pp) {
        const int mt = mt0 + pp;
#pragma unroll
        for (int jj = 0; jj < 8; ++jj) {
            const int brw = mt * 16 + jj + 8 * hv;      // batch row
            const float gv = G[((size_t)brw * T + t) * (4 * H) + gate * H + n0 + r];
            zb[gate][brw][r] = (pp ? acc1[jj] : acc0[jj]) + gv;
        }
    }
    __syncthreads();

#pragma unroll
    for (int i = 0; i < 4; ++i) {
        const int e = threadIdx.x + 256 * i;            // 0..1023
        const int b = e >> 4;
        const int n = e & 15;
        const float zi = zb[0][b][n], zf = zb[1][b][n];
        const float zg = zb[2][b][n], zo = zb[3][b][n];
        const float ig = 1.0f / (1.0f + expf(-zi));
        const float fg = 1.0f / (1.0f + expf(-zf));
        const float gg = tanhf(zg);
        const float og = 1.0f / (1.0f + expf(-zo));
        const int col = n0 + n;
        const size_t si = (size_t)b * H + col;
        const float cn = fg * cstate[si] + ig * gg;
        const float hn = og * tanhf(cn);
        cstate[si] = cn;
        hnext[si]  = f2bf(hn);
        hseq[((size_t)b * T + t) * H + col] = f2bf(hn);
        if (t == T - 1) { c_out[si] = cn; h_out[si] = hn; }
    }
}

// ---------------------------------------------------------------------------
// Helpers: fp32 -> bf16 copy, fp32 -> bf16 transpose (LDS tile), state init.
// ---------------------------------------------------------------------------
__global__ void f32_to_bf16_k(const float* __restrict__ in,
                              __bf16* __restrict__ out, long n) {
    long i = (long)blockIdx.x * blockDim.x + threadIdx.x;
    long stride = (long)gridDim.x * blockDim.x;
    for (; i < n; i += stride) out[i] = f2bf(in[i]);
}

__global__ void transpose_bf16_k(const float* __restrict__ W,
                                 __bf16* __restrict__ Wt, int K, int N) {
    __shared__ float tile[16][17];
    const int n_in = blockIdx.x * 16 + threadIdx.x;
    const int k_in = blockIdx.y * 16 + threadIdx.y;
    tile[threadIdx.y][threadIdx.x] =
        (k_in < K && n_in < N) ? W[(size_t)k_in * N + n_in] : 0.0f;
    __syncthreads();
    const int n_out = blockIdx.x * 16 + threadIdx.y;
    const int k_out = blockIdx.y * 16 + threadIdx.x;
    if (n_out < N && k_out < K)
        Wt[(size_t)n_out * K + k_out] = f2bf(tile[threadIdx.x][threadIdx.y]);
}

__global__ void init_states_k(const float* __restrict__ c0,
                              const float* __restrict__ h0,
                              float* __restrict__ cstate,
                              __bf16* __restrict__ h_l0,
                              __bf16* __restrict__ h_l1, int BH) {
    int i = blockIdx.x * blockDim.x + threadIdx.x;
    if (i >= 2 * BH) return;
    cstate[i] = c0[i];
    const __bf16 hb = f2bf(h0[i]);
    if (i < BH) h_l0[i] = hb; else h_l1[i - BH] = hb;
}

// ---------------------------------------------------------------------------
extern "C" void kernel_launch(void* const* d_in, const int* in_sizes, int n_in,
                              void* d_out, int out_size, void* d_ws, size_t ws_size,
                              hipStream_t stream) {
    (void)in_sizes; (void)n_in; (void)out_size; (void)ws_size;
    const int B = 64, T = 128, F = 512, H = 1024, P = 1024, Adim = 32;
    const int M = B * T;                       // 8192 rows

    const float* x       = (const float*)d_in[0];
    const float* c0      = (const float*)d_in[1];
    const float* h0      = (const float*)d_in[2];
    const float* W_pre   = (const float*)d_in[3];
    const float* b_pre   = (const float*)d_in[4];
    const float* ln_s    = (const float*)d_in[5];
    const float* ln_b    = (const float*)d_in[6];
    const float* Wi0     = (const float*)d_in[7];
    const float* Wh0     = (const float*)d_in[8];
    const float* b0      = (const float*)d_in[9];
    const float* Wi1     = (const float*)d_in[10];
    const float* Wh1     = (const float*)d_in[11];
    const float* b1      = (const float*)d_in[12];
    const float* W_post  = (const float*)d_in[13];
    const float* b_post  = (const float*)d_in[14];
    const float* W_out   = (const float*)d_in[15];
    const float* b_out   = (const float*)d_in[16];

    // workspace carve-out (256B aligned)
    char* w = (char*)d_ws;
    auto carve = [&](size_t bytes) {
        char* p = w; w += (bytes + 255) & ~(size_t)255; return p;
    };
    float*  Gbuf   = (float*) carve((size_t)M * 4 * H * sizeof(float)); // 134MB, reused
    __bf16* Xbf    = (__bf16*)carve((size_t)M * F * 2);
    __bf16* Xp     = (__bf16*)carve((size_t)M * P * 2);
    __bf16* hseq0  = (__bf16*)carve((size_t)M * H * 2);
    __bf16* hseq1  = (__bf16*)carve((size_t)M * H * 2);
    __bf16* postb  = (__bf16*)carve((size_t)M * P * 2);
    __bf16* WpreT  = (__bf16*)carve((size_t)P * F * 2);
    __bf16* Wi0T   = (__bf16*)carve((size_t)4 * H * P * 2);
    __bf16* Wh0T   = (__bf16*)carve((size_t)4 * H * H * 2);
    __bf16* Wi1T   = (__bf16*)carve((size_t)4 * H * H * 2);
    __bf16* Wh1T   = (__bf16*)carve((size_t)4 * H * H * 2);
    __bf16* WpostT = (__bf16*)carve((size_t)P * H * 2);
    __bf16* WoutT  = (__bf16*)carve((size_t)Adim * P * 2);
    float*  cstate = (float*) carve((size_t)2 * B * H * sizeof(float));
    __bf16* hbuf   = (__bf16*)carve((size_t)2 * 2 * B * H * 2); // [layer][sel][B*H]

    float* dout   = (float*)d_out;
    float* c_out0 = dout;                    // stack([c1,c2]) : [2][B][H]
    float* c_out1 = dout + (size_t)B * H;
    float* h_out0 = dout + (size_t)2 * B * H;
    float* h_out1 = dout + (size_t)3 * B * H;
    float* logits = dout + (size_t)4 * B * H;

    // 1) input + weight conversion (weights stored transposed, bf16)
    f32_to_bf16_k<<<1024, 256, 0, stream>>>(x, Xbf, (long)M * F);
    {
        dim3 blk(16, 16);
        transpose_bf16_k<<<dim3(P / 16, F / 16),        blk, 0, stream>>>(W_pre,  WpreT,  F, P);
        transpose_bf16_k<<<dim3(4 * H / 16, P / 16),    blk, 0, stream>>>(Wi0,    Wi0T,   P, 4 * H);
        transpose_bf16_k<<<dim3(4 * H / 16, H / 16),    blk, 0, stream>>>(Wh0,    Wh0T,   H, 4 * H);
        transpose_bf16_k<<<dim3(4 * H / 16, H / 16),    blk, 0, stream>>>(Wi1,    Wi1T,   H, 4 * H);
        transpose_bf16_k<<<dim3(4 * H / 16, H / 16),    blk, 0, stream>>>(Wh1,    Wh1T,   H, 4 * H);
        transpose_bf16_k<<<dim3(P / 16, H / 16),        blk, 0, stream>>>(W_post, WpostT, H, P);
        transpose_bf16_k<<<dim3((Adim + 15) / 16, P / 16), blk, 0, stream>>>(W_out, WoutT, P, Adim);
    }
    init_states_k<<<(2 * B * H + 255) / 256, 256, 0, stream>>>(
        c0, h0, cstate, hbuf, hbuf + (size_t)2 * B * H, B * H);

    // 2) pre: z = x @ W_pre + b_pre  (fp32 into Gbuf, reused as scratch)
    gemm_bf16_k<false, false><<<dim3(P / 64, M / 128), 256, 0, stream>>>(
        Xbf, WpreT, b_pre, Gbuf, M, P, F);
    // 3) Xp = relu(LN(z))
    ln_relu_k<<<M, 256, 0, stream>>>(Gbuf, ln_s, ln_b, Xp);

    // 4) G = Xp @ Wi0 + b0  (all timesteps at once)
    gemm_bf16_k<false, false><<<dim3(4 * H / 64, M / 128), 256, 0, stream>>>(
        Xp, Wi0T, b0, Gbuf, M, 4 * H, P);
    // 5) layer-0 scan
    for (int t = 0; t < T; ++t) {
        __bf16* hp = hbuf + (size_t)(t & 1) * B * H;
        __bf16* hn = hbuf + (size_t)((t + 1) & 1) * B * H;
        lstm_step_k<<<H / 16, 256, 0, stream>>>(
            Gbuf, Wh0T, cstate, hp, hn, hseq0, c_out0, h_out0, t, B, T, H);
    }

    // 6) G = hseq0 @ Wi1 + b1
    gemm_bf16_k<false, false><<<dim3(4 * H / 64, M / 128), 256, 0, stream>>>(
        hseq0, Wi1T, b1, Gbuf, M, 4 * H, H);
    // 7) layer-1 scan
    __bf16* hbuf1 = hbuf + (size_t)2 * B * H;
    float*  cst1  = cstate + (size_t)B * H;
    for (int t = 0; t < T; ++t) {
        __bf16* hp = hbuf1 + (size_t)(t & 1) * B * H;
        __bf16* hn = hbuf1 + (size_t)((t + 1) & 1) * B * H;
        lstm_step_k<<<H / 16, 256, 0, stream>>>(
            Gbuf, Wh1T, cst1, hp, hn, hseq1, c_out1, h_out1, t, B, T, H);
    }

    // 8) post: relu(hseq1 @ W_post + b_post) -> bf16
    gemm_bf16_k<true, true><<<dim3(P / 64, M / 128), 256, 0, stream>>>(
        hseq1, WpostT, b_post, postb, M, P, H);
    // 9) logits = post @ W_out + b_out -> fp32 into d_out
    gemm_bf16_k<false, false><<<dim3(1, M / 128), 256, 0, stream>>>(
        postb, WoutT, b_out, logits, M, Adim, P);
}
